// SparseContextAttention_83451214561953
// MI455X (gfx1250) — matrix operand order repaired and checked by reading, not compiled, not run on hardware
//
#include <hip/hip_runtime.h>
#include <hip/hip_bf16.h>
#include <stdint.h>

#define BB 4      // batch
#define LQ 1024   // queries
#define LC 4096   // context length
#define DM 512    // d_model
#define NH 8      // heads
#define DH 64     // head dim
#define TK 32     // top-k

#define BM 256    // block tile rows (8 waves x 32)
#define BN 64     // block tile cols
#define BK 32     // K step (one WMMA contraction)
#define LDSTR 40  // padded LDS stride in halfs per column (80B -> conflict-free, 16B aligned)

typedef __attribute__((ext_vector_type(16))) __bf16 v16bf;
typedef __attribute__((ext_vector_type(8)))  float  v8f;

union Frag { v16bf v; uint4 u[2]; };

__device__ __forceinline__ uint16_t f2bf(float x) {
  uint32_t u = __float_as_uint(x);
  uint32_t r = u + 0x7FFFu + ((u >> 16) & 1u);   // round-to-nearest-even
  return (uint16_t)(r >> 16);
}

// ---------------- fp32 -> bf16 conversion ----------------
__global__ void cvt_f32_bf16_kernel(const float* __restrict__ in,
                                    uint16_t* __restrict__ out, int n) {
  int i = blockIdx.x * blockDim.x + threadIdx.x;
  if (i < n) out[i] = f2bf(in[i]);
}

// ---------------- WMMA GEMM: C[M,N] = A[M,512] * W[N,512]^T + bias ----------
// 256-thread block = 8 waves; block tile 256x64; wave tile 32x64 (2 Mx4 N wmma
// tiles, 8 fp32 accumulators). B (64x32 per K-step) staged in double-buffered
// LDS and shared by all 8 waves. Global loads for step k+1 are issued before
// the 8 WMMAs of step k; the LDS commit happens after them, so the load-wait
// is covered by matrix math. A fragments are register double-buffered. All
// four B fragments are batch-preloaded so one dscnt wait covers 8 WMMAs.
template<bool BF16_OUT>
__global__ void __launch_bounds__(256)
gemm_nt_wmma(const uint16_t* __restrict__ A,
             const uint16_t* __restrict__ W,
             const float* __restrict__ bias,
             void* __restrict__ out, int N) {
  __shared__ uint16_t ldsB[2 * BN * LDSTR];

  const int tid  = threadIdx.x;
  const int wave = tid >> 5;
  const int lane = tid & 31;
  const int r    = lane & 15;
  const int hi   = lane >> 4;           // lane half: selects K sub-range

  const int m0 = blockIdx.x * BM + wave * 32;
  const int n0 = blockIdx.y * BN;

  // Staging: each thread moves one b128 of W per K step.
  const int scol = tid >> 2;            // 0..63 (column within block tile)
  const int skp  = (tid & 3) * 8;       // 0,8,16,24 (halfs within K step)
  const uint16_t* sW = W + (size_t)(n0 + scol) * DM + skp;
  uint16_t*       sL = &ldsB[scol * LDSTR + skp];

  // A-frag layout (16x32 bf16): lane half 0 -> K {0..7,16..23}; half 1 -> {8..15,24..31}
  const uint16_t* Arow0 = A + (size_t)(m0 + r) * DM + hi * 8;
  const uint16_t* Arow1 = Arow0 + (size_t)16 * DM;
  // B-frag read base: lane = column, lane half 0 -> K {0..7,16..23}; half 1 -> {8..15,24..31}
  const uint16_t* fbp = &ldsB[(lane & 15) * LDSTR + hi * 8];

  const v8f vz = {0.f,0.f,0.f,0.f,0.f,0.f,0.f,0.f};
  v8f acc[2][4] = {{vz,vz,vz,vz},{vz,vz,vz,vz}};

  Frag fa[2][2];
  // prologue: K step 0 into register buffer 0 / LDS buffer 0
  fa[0][0].u[0] = *(const uint4*)(Arow0);
  fa[0][0].u[1] = *(const uint4*)(Arow0 + 16);
  fa[0][1].u[0] = *(const uint4*)(Arow1);
  fa[0][1].u[1] = *(const uint4*)(Arow1 + 16);
  *(uint4*)sL = *(const uint4*)sW;
  __syncthreads();

#pragma unroll
  for (int kk = 0; kk < DM / BK; ++kk) {
    const int k   = kk * BK;
    const int cur = kk & 1;
    const int nxt = cur ^ 1;

    uint4 stg;
    if (kk < DM / BK - 1) {
      // issue next-step global loads now; consume them after the WMMAs
      stg = *(const uint4*)(sW + k + BK);
      __builtin_prefetch(sW + k + 2 * BK, 0, 3);     // WGP-scope global_prefetch_b8
      fa[nxt][0].u[0] = *(const uint4*)(Arow0 + k + BK);
      fa[nxt][0].u[1] = *(const uint4*)(Arow0 + k + BK + 16);
      fa[nxt][1].u[0] = *(const uint4*)(Arow1 + k + BK);
      fa[nxt][1].u[1] = *(const uint4*)(Arow1 + k + BK + 16);
    }

    // batch-preload all four B fragments (one ds clause, one dscnt wait)
    const uint16_t* fb0 = fbp + cur * (BN * LDSTR);
    Frag wf[4];
#pragma unroll
    for (int j = 0; j < 4; ++j) {
      wf[j].u[0] = *(const uint4*)(fb0 + j * 16 * LDSTR);
      wf[j].u[1] = *(const uint4*)(fb0 + j * 16 * LDSTR + 16);
    }
#pragma unroll
    for (int j = 0; j < 4; ++j) {
      acc[0][j] = __builtin_amdgcn_wmma_f32_16x16x32_bf16(
          false, fa[cur][0].v, false, wf[j].v, (short)0, acc[0][j], false, false);
      acc[1][j] = __builtin_amdgcn_wmma_f32_16x16x32_bf16(
          false, fa[cur][1].v, false, wf[j].v, (short)0, acc[1][j], false, false);
    }

    if (kk < DM / BK - 1) {
      *(uint4*)(sL + nxt * (BN * LDSTR)) = stg;      // ds_store_b128 (other buffer)
    }
    __syncthreads();
  }

  // C/D layout: VGPR i -> row (+8 for upper lane half), lane&15 -> column
  const int col0 = n0 + r;
#pragma unroll
  for (int j = 0; j < 4; ++j) {
    const int col = col0 + j * 16;
    const float bv = bias ? bias[col] : 0.0f;
#pragma unroll
    for (int mi = 0; mi < 2; ++mi) {
#pragma unroll
      for (int i = 0; i < 8; ++i) {
        const int row = m0 + mi * 16 + i + hi * 8;
        const float val = acc[mi][j][i] + bv;
        if constexpr (BF16_OUT)
          ((uint16_t*)out)[(size_t)row * N + col] = f2bf(val);
        else
          ((float*)out)[(size_t)row * N + col] = val;
      }
    }
  }
}

// ---------------- top-32 of each 4096-wide sim row --------------------------
__global__ void topk_kernel(const float* __restrict__ sim, int* __restrict__ idx) {
  __shared__ float sv[LC];
  __shared__ float rv[256];
  __shared__ int   rp[256];
  const int q   = blockIdx.x;
  const int tid = threadIdx.x;
  const float* row = sim + (size_t)q * LC;
  for (int j = tid; j < LC; j += 256) sv[j] = row[j];
  __syncthreads();
  int* op = idx + q * TK;
  for (int t = 0; t < TK; ++t) {
    float best = -INFINITY; int bp = LC;
    for (int j = tid; j < LC; j += 256) {
      float v = sv[j];
      if (v > best) { best = v; bp = j; }
    }
    rv[tid] = best; rp[tid] = bp;
    __syncthreads();
    for (int s = 128; s > 0; s >>= 1) {
      if (tid < s) {
        float ov = rv[tid + s]; int opos = rp[tid + s];
        if (ov > rv[tid] || (ov == rv[tid] && opos < rp[tid])) {
          rv[tid] = ov; rp[tid] = opos;
        }
      }
      __syncthreads();
    }
    if (tid == 0) { op[t] = rp[0]; sv[rp[0]] = -INFINITY; }
    __syncthreads();
  }
}

// ---------------- attention core: one wave per (b,q,h) ----------------------
__global__ void attn_kernel(const float* __restrict__ qv,   // [B*Lq, 512]
                            const float* __restrict__ kc,   // [B*Lc, 512]
                            const float* __restrict__ vc,   // [B*Lc, 512]
                            const int*   __restrict__ idx,  // [B*Lq, 32]
                            uint16_t* __restrict__ out_bf)  // [B*Lq, 512] bf16
{
  const int gwave = (blockIdx.x * blockDim.x + threadIdx.x) >> 5;
  const int lane  = threadIdx.x & 31;
  const int h = gwave % NH;
  const int q = (gwave / NH) % LQ;
  const int b = gwave / (NH * LQ);
  const int row = b * LQ + q;

  const float* qp = qv + (size_t)row * DM + h * DH;      // uniform across wave
  const int myIdx = idx[(size_t)row * TK + lane];        // this lane's key
  const float* kp = kc + (size_t)(b * LC + myIdx) * DM + h * DH;

  float s = 0.f;
#pragma unroll
  for (int d = 0; d < DH; d += 4) {
    float4 k4 = *(const float4*)(kp + d);
    float4 q4 = *(const float4*)(qp + d);
    s += k4.x * q4.x + k4.y * q4.y + k4.z * q4.z + k4.w * q4.w;
  }
  s *= 0.125f;  // 1/sqrt(64)

  float m = s;
  for (int o = 16; o > 0; o >>= 1) m = fmaxf(m, __shfl_xor(m, o, 32));
  const float e = __expf(s - m);
  float sum = e;
  for (int o = 16; o > 0; o >>= 1) sum += __shfl_xor(sum, o, 32);
  const float a = e / sum;

  float acc0 = 0.f, acc1 = 0.f;
  for (int k2 = 0; k2 < TK; ++k2) {
    const float ak = __shfl(a, k2, 32);
    const int   ki = __shfl(myIdx, k2, 32);
    const float2 v2 = *(const float2*)(vc + (size_t)(b * LC + ki) * DM + h * DH + lane * 2);
    acc0 += ak * v2.x;
    acc1 += ak * v2.y;
  }
  const size_t o = (size_t)row * DM + h * DH + lane * 2;
  out_bf[o]     = f2bf(acc0);
  out_bf[o + 1] = f2bf(acc1);
}

// ---------------- launcher --------------------------------------------------
extern "C" void kernel_launch(void* const* d_in, const int* in_sizes, int n_in,
                              void* d_out, int out_size, void* d_ws, size_t ws_size,
                              hipStream_t stream) {
  (void)in_sizes; (void)n_in; (void)out_size; (void)ws_size;
  const float* chunk   = (const float*)d_in[0];
  const float* context = (const float*)d_in[1];
  const float* W_sim   = (const float*)d_in[2];
  const float* b_sim   = (const float*)d_in[3];
  const float* Wq      = (const float*)d_in[4];
  const float* bq      = (const float*)d_in[5];
  const float* Wk      = (const float*)d_in[6];
  const float* bk      = (const float*)d_in[7];
  const float* Wv      = (const float*)d_in[8];
  const float* bv      = (const float*)d_in[9];
  const float* Wo      = (const float*)d_in[10];
  const float* bo      = (const float*)d_in[11];

  char* p = (char*)d_ws;
  auto carve = [&](size_t bytes) -> char* {
    char* q = p;
    p += (bytes + 255) & ~size_t(255);
    return q;
  };
  uint16_t* chunk_bf  = (uint16_t*)carve((size_t)BB * LQ * DM * 2);
  uint16_t* ctx_bf    = (uint16_t*)carve((size_t)BB * LC * DM * 2);
  uint16_t* wsim_bf   = (uint16_t*)carve((size_t)DM * DM * 2);
  uint16_t* wq_bf     = (uint16_t*)carve((size_t)DM * DM * 2);
  uint16_t* wk_bf     = (uint16_t*)carve((size_t)DM * DM * 2);
  uint16_t* wv_bf     = (uint16_t*)carve((size_t)DM * DM * 2);
  uint16_t* wo_bf     = (uint16_t*)carve((size_t)DM * DM * 2);
  uint16_t* simctx_bf = (uint16_t*)carve((size_t)BB * LC * DM * 2);
  float*    kc        = (float*)carve((size_t)BB * LC * DM * 4);
  float*    vc        = (float*)carve((size_t)BB * LC * DM * 4);
  float*    qv        = (float*)carve((size_t)BB * LQ * DM * 4);
  float*    sim       = (float*)carve((size_t)LQ * LC * 4);   // reused per batch
  int*      idx       = (int*)carve((size_t)BB * LQ * TK * 4);
  uint16_t* attn_bf   = (uint16_t*)carve((size_t)BB * LQ * DM * 2);

  auto cvt = [&](const float* src, uint16_t* dst, int n) {
    cvt_f32_bf16_kernel<<<(n + 255) / 256, 256, 0, stream>>>(src, dst, n);
  };
  cvt(chunk,   chunk_bf, BB * LQ * DM);
  cvt(context, ctx_bf,   BB * LC * DM);
  cvt(W_sim,   wsim_bf,  DM * DM);
  cvt(Wq,      wq_bf,    DM * DM);
  cvt(Wk,      wk_bf,    DM * DM);
  cvt(Wv,      wv_bf,    DM * DM);
  cvt(Wo,      wo_bf,    DM * DM);

  // Whole-context projections (8x fewer FLOPs than per-gather projection).
  gemm_nt_wmma<true ><<<dim3(BB * LC / BM, DM / BN), 256, 0, stream>>>(ctx_bf,   wsim_bf, b_sim, simctx_bf, DM);
  gemm_nt_wmma<false><<<dim3(BB * LC / BM, DM / BN), 256, 0, stream>>>(ctx_bf,   wk_bf,   bk,    kc,        DM);
  gemm_nt_wmma<false><<<dim3(BB * LC / BM, DM / BN), 256, 0, stream>>>(ctx_bf,   wv_bf,   bv,    vc,        DM);
  gemm_nt_wmma<false><<<dim3(BB * LQ / BM, DM / BN), 256, 0, stream>>>(chunk_bf, wq_bf,   bq,    qv,        DM);

  // Per-batch: sim = chunk . sim_ctx^T (16.8 MB ws reuse), then top-32.
  for (int b = 0; b < BB; ++b) {
    gemm_nt_wmma<false><<<dim3(LQ / BM, LC / BN), 256, 0, stream>>>(
        chunk_bf + (size_t)b * LQ * DM,
        simctx_bf + (size_t)b * LC * DM,
        nullptr, sim, LC);
    topk_kernel<<<LQ, 256, 0, stream>>>(sim, idx + (size_t)b * LQ * TK);
  }

  // Attention core: B*Lq*H = 32768 waves, 8 waves per block.
  attn_kernel<<<BB * LQ * NH / 8, 256, 0, stream>>>(qv, kc, vc, idx, attn_bf);

  // Output projection straight into d_out (fp32 + bias).
  gemm_nt_wmma<false><<<dim3(BB * LQ / BM, DM / BN), 256, 0, stream>>>(attn_bf, wo_bf, bo, (float*)d_out, DM);
}
